// GrowableTwistorLNN_9345848836662
// MI455X (gfx1250) — matrix operand order repaired and checked by reading, not compiled, not run on hardware
//
#include <hip/hip_runtime.h>
#include <hip/hip_bf16.h>

typedef _Float16 h16;
typedef __attribute__((ext_vector_type(16))) _Float16 v16h;
typedef __attribute__((ext_vector_type(8)))  _Float16 v8h;
typedef __attribute__((ext_vector_type(8)))  float    v8f;
typedef __attribute__((ext_vector_type(4)))  float    v4f;

#define T_STEPS 1024
#define B_SZ    512
#define IN_SZ   64
#define H_SZ    128
#define AS      136   // padded LDS row stride (halfs) for activation tiles (272B -> conflict-free b128)
#define XS      72    // padded LDS row stride (halfs) for x tile

#define LOG2E 1.4426950408889634f

// Fast hardware transcendentals (v_rcp_f32 / v_exp_f32 / v_sqrt_f32, ~1 ulp).
// Safe here: tanh/sigmoid saturate, tau is clamped to [0.01, 1], z to +-100.
__device__ __forceinline__ float frcp(float x)  { return __builtin_amdgcn_rcpf(x); }
__device__ __forceinline__ float fexp2(float x) { return __builtin_amdgcn_exp2f(x); }
__device__ __forceinline__ float fsqrt(float x) { return __builtin_amdgcn_sqrtf(x); }
__device__ __forceinline__ float fsigm(float x) { return frcp(1.0f + fexp2(-x * LOG2E)); }
__device__ __forceinline__ float ftanh(float x) {
  // tanh(x) = 2*sigmoid(2x) - 1; exp2 saturates cleanly at the tails.
  return 2.0f * frcp(1.0f + fexp2(-2.0f * LOG2E * x)) - 1.0f;
}

// ---------------------------------------------------------------------------
// Prologue: fold sigmoid mask into recurrent weights, convert all GEMM
// weights to f16 row-major [N][K] (exactly the WMMA B operand for x @ W.T).
// ---------------------------------------------------------------------------
__global__ void twistor_prep(const float* __restrict__ Wre, const float* __restrict__ Wim,
                             const float* __restrict__ mre, const float* __restrict__ mim,
                             const float* __restrict__ Wtau, const float* __restrict__ U,
                             h16* __restrict__ WrF, h16* __restrict__ WiF,
                             h16* __restrict__ WtF, h16* __restrict__ UF) {
  int i = blockIdx.x * 256 + threadIdx.x;
  if (i < H_SZ * H_SZ) {
    WrF[i] = (h16)(Wre[i] * fsigm(mre[i]));
    WiF[i] = (h16)(Wim[i] * fsigm(mim[i]));
    WtF[i] = (h16)(Wtau[i]);
  }
  if (i < H_SZ * IN_SZ) UF[i] = (h16)U[i];
}

// A-fragment (16x32 f16) loader from LDS, contiguous K order, same as the B fragments:
// lane<16 : row=lane,    K = kt*32 + {0..15}
// lane>=16: row=lane-16, K = kt*32 + {16..31}
__device__ __forceinline__ v16h load_afrag(const h16* base, int stride, int kt, int nl, int hi) {
  const h16* p = base + nl * stride + kt * 32 + hi * 16;
  v8h a = *(const v8h*)(p);
  v8h b = *(const v8h*)(p + 8);
  v16h r;
#pragma unroll
  for (int i = 0; i < 8; ++i) { r[i] = a[i]; r[8 + i] = b[i]; }
  return r;
}

#define WMMA(A, B, C) \
  __builtin_amdgcn_wmma_f32_16x16x32_f16(false, (A), false, (B), (short)0, (C), false, false)

// ---------------------------------------------------------------------------
// Main recurrent kernel: one workgroup = 16 batch rows, 8 waves; wave w owns
// hidden columns [16w, 16w+16). Entire T=1024 loop stays in-kernel; all
// weight B-fragments stay resident in VGPRs for the whole loop.
// ---------------------------------------------------------------------------
__global__ __launch_bounds__(256, 1)
void twistor_main(const float* __restrict__ x,
                  const h16* __restrict__ WrF, const h16* __restrict__ WiF,
                  const h16* __restrict__ WtF, const h16* __restrict__ UF,
                  const float* __restrict__ U_b, const float* __restrict__ Wtau_b,
                  const float* __restrict__ tau_bias, const float* __restrict__ b_real,
                  const float* __restrict__ b_imag, const float* __restrict__ out_w,
                  const float* __restrict__ out_b, float* __restrict__ y) {
  __shared__ h16 A_tr[16 * AS];
  __shared__ h16 A_ti[16 * AS];
  __shared__ h16 A_zm[16 * AS];
  __shared__ h16 XA[16 * XS];
  __shared__ float ypart[8 * 16];

  const int tid  = threadIdx.x;
  const int lane = tid & 31;
  const int wv   = tid >> 5;          // wave id 0..7 == hidden column tile
  const int nl   = lane & 15;         // column within tile
  const int hi   = (lane >> 4) & 1;   // lane half
  const int n    = wv * 16 + nl;      // hidden column owned by this lane
  const int batch0 = blockIdx.x * 16;

  // ---- resident B fragments (B[k][n] = W[n][k], contiguous 16 K per lane) ----
  v16h bWr[4], bWi[4], bWt[4], bU[2];
  {
    const int ks = hi * 16;
#pragma unroll
    for (int kt = 0; kt < 4; ++kt) {
      bWr[kt] = *(const v16h*)(WrF + n * H_SZ + kt * 32 + ks);
      bWi[kt] = *(const v16h*)(WiF + n * H_SZ + kt * 32 + ks);
      bWt[kt] = *(const v16h*)(WtF + n * H_SZ + kt * 32 + ks);
    }
#pragma unroll
    for (int kt = 0; kt < 2; ++kt)
      bU[kt] = *(const v16h*)(UF + n * IN_SZ + kt * 32 + ks);
  }

  // ---- per-lane column constants ----
  const float ub   = U_b[n];
  const float cR   = ub + b_real[n];
  const float cI   = ub + b_imag[n];
  const float wtb  = Wtau_b[n];
  const float tb   = tau_bias[n];
  const float wout = out_w[n];
  const float ob   = out_b[0];

  // ---- state z = 0 (C-layout: VGPR v -> row v + 8*hi, col n) ----
  v8f zr = {}; v8f zi = {};

  // zero activation tiles (z0 = 0 -> tanh = 0, |z| = 0), stage x_0
  for (int i = tid; i < 16 * AS; i += 256) {
    A_tr[i] = (h16)0.0f; A_ti[i] = (h16)0.0f; A_zm[i] = (h16)0.0f;
  }
  {
    int e = tid * 4;
    int row = e >> 6, col = e & 63;
    v4f v = *(const v4f*)(x + ((size_t)(batch0 + row)) * IN_SZ + col);
    h16* d = XA + row * XS + col;
    d[0] = (h16)v.x; d[1] = (h16)v.y; d[2] = (h16)v.z; d[3] = (h16)v.w;
  }
  __syncthreads();

  for (int t = 0; t < T_STEPS; ++t) {
    // ---- GEMMs: Ux shared by both dz chains; three interleaved chains ----
    v8f accU = {};
#pragma unroll
    for (int kt = 0; kt < 2; ++kt) {
      v16h ax = load_afrag(XA, XS, kt, nl, hi);
      accU = WMMA(ax, bU[kt], accU);
    }
    v8f ar = accU, ai = accU, at = {};
#pragma unroll
    for (int kt = 0; kt < 4; ++kt) {
      v16h a1 = load_afrag(A_tr, AS, kt, nl, hi);
      ar = WMMA(a1, bWr[kt], ar);
      v16h a2 = load_afrag(A_ti, AS, kt, nl, hi);
      ai = WMMA(a2, bWi[kt], ai);
      v16h a3 = load_afrag(A_zm, AS, kt, nl, hi);
      at = WMMA(a3, bWt[kt], at);
    }

    // ---- elementwise liquid update + output partials (fast VALU path) ----
    float trv[8], tiv[8], zmv[8], yp[8];
#pragma unroll
    for (int v = 0; v < 8; ++v) {
      float tau = fsigm(at[v] + wtb) + tb;
      tau = fminf(fmaxf(tau, 0.01f), 1.0f) + 1e-6f;
      float rt = frcp(tau);                 // one rcp replaces two divisions
      float zrv = zr[v], ziv = zi[v];
      float dr = (-zrv + ar[v] + cR) * rt;
      float di = (-ziv + ai[v] + cI) * rt;
      dr = fminf(fmaxf(dr, -10.0f), 10.0f);
      di = fminf(fmaxf(di, -10.0f), 10.0f);
      zrv = fminf(fmaxf(zrv + 0.1f * dr, -100.0f), 100.0f);
      ziv = fminf(fmaxf(ziv + 0.1f * di, -100.0f), 100.0f);
      zr[v] = zrv; zi[v] = ziv;
      trv[v] = ftanh(zrv);
      tiv[v] = ftanh(ziv);
      zmv[v] = fsqrt(zrv * zrv + ziv * ziv);
      // y_t partial: reduce this wave's 16 columns within each lane half
      float p = zrv * wout;
      p += __shfl_xor(p, 1, 32);
      p += __shfl_xor(p, 2, 32);
      p += __shfl_xor(p, 4, 32);
      p += __shfl_xor(p, 8, 32);
      yp[v] = p;
    }

    __syncthreads();  // all WMMA LDS reads of this step are complete

    // ---- publish next-step A operands + y partials ----
    {
      const int row0 = hi * 8;
#pragma unroll
      for (int v = 0; v < 8; ++v) {
        int off = (row0 + v) * AS + n;
        A_tr[off] = (h16)trv[v];
        A_ti[off] = (h16)tiv[v];
        A_zm[off] = (h16)zmv[v];
      }
      if (nl == 0) {
#pragma unroll
        for (int v = 0; v < 8; ++v) ypart[wv * 16 + row0 + v] = yp[v];
      }
    }
    // ---- stage x_{t+1}, prefetch x_{t+2} ----
    if (t + 1 < T_STEPS) {
      int e = tid * 4;
      int row = e >> 6, col = e & 63;
      const float* xp = x + ((size_t)(t + 1) * B_SZ + batch0 + row) * IN_SZ + col;
      v4f v = *(const v4f*)xp;
      h16* d = XA + row * XS + col;
      d[0] = (h16)v.x; d[1] = (h16)v.y; d[2] = (h16)v.z; d[3] = (h16)v.w;
      if (t + 2 < T_STEPS)
        __builtin_prefetch(x + ((size_t)(t + 2) * B_SZ + batch0 + row) * IN_SZ + col, 0, 1);
    }
    __syncthreads();

    // ---- final y_t for this batch tile ----
    if (tid < 16) {
      float s = ob;
#pragma unroll
      for (int w8 = 0; w8 < 8; ++w8) s += ypart[w8 * 16 + tid];
      y[(size_t)t * B_SZ + batch0 + tid] = s;
    }
  }
}

extern "C" void kernel_launch(void* const* d_in, const int* in_sizes, int n_in,
                              void* d_out, int out_size, void* d_ws, size_t ws_size,
                              hipStream_t stream) {
  (void)in_sizes; (void)n_in; (void)out_size; (void)ws_size;
  const float* x      = (const float*)d_in[0];
  const float* W_real = (const float*)d_in[1];
  const float* W_imag = (const float*)d_in[2];
  const float* m_real = (const float*)d_in[3];
  const float* m_imag = (const float*)d_in[4];
  const float* U_w    = (const float*)d_in[5];
  const float* U_b    = (const float*)d_in[6];
  const float* Wtau_w = (const float*)d_in[7];
  const float* Wtau_b = (const float*)d_in[8];
  const float* tau_b  = (const float*)d_in[9];
  const float* b_real = (const float*)d_in[10];
  const float* b_imag = (const float*)d_in[11];
  const float* out_w  = (const float*)d_in[12];
  const float* out_b  = (const float*)d_in[13];
  float* y = (float*)d_out;

  // workspace: f16 weight images (112 KB total)
  h16* WrF = (h16*)d_ws;
  h16* WiF = WrF + H_SZ * H_SZ;
  h16* WtF = WiF + H_SZ * H_SZ;
  h16* UF  = WtF + H_SZ * H_SZ;

  twistor_prep<<<(H_SZ * H_SZ + 255) / 256, 256, 0, stream>>>(
      W_real, W_imag, m_real, m_imag, Wtau_w, U_w, WrF, WiF, WtF, UF);
  twistor_main<<<B_SZ / 16, 256, 0, stream>>>(
      x, WrF, WiF, WtF, UF, U_b, Wtau_b, tau_b, b_real, b_imag, out_w, out_b, y);
}
